// MultiHeadAttention_60722247631281
// MI455X (gfx1250) — matrix operand order repaired and checked by reading, not compiled, not run on hardware
//
#include <hip/hip_runtime.h>
#include <hip/hip_bf16.h>

typedef __attribute__((ext_vector_type(16))) _Float16 v16h;
typedef __attribute__((ext_vector_type(8)))  _Float16 v8h;
typedef __attribute__((ext_vector_type(8)))  float    v8f;

#define NUM_B  8
#define SEQ_N  1025
#define NPAD   1056          // multiple of 32 (key/M2 tiles) and 16 (M tiles)
#define DMODEL 768
#define NHEAD  12
#define HD     64
#define MTILES  (NPAD / 16)  // 66  (flash q-tiles)
#define M2TILES (NPAD / 32)  // 33  (gemm m-tiles, 32 rows per wave)
#define KTILES  (NPAD / 32)  // 33  (flash key tiles)

// ---------------------------------------------------------------------------
// WMMA fragment loaders (wave32; A/B use this file's contiguous k order, not the ISA table order)
// A (16x32, MxK): lane = 16g + m ; halves i   -> K=16g+i
// B (32x16, KxN): lane = 16g + n ; halves i   -> K=16g+i
// C (16x16):      lane = 16g + n ; vgpr j     -> M=j+8g
// ---------------------------------------------------------------------------
static __device__ __forceinline__ v16h load_a16(const _Float16* A,
                                                size_t rowoff, int k0, int g) {
  const _Float16* p = A + rowoff + (size_t)k0 + 16 * g;
  v8h lo = *(const v8h*)(p);
  v8h hi = *(const v8h*)(p + 8);
  v16h r;
#pragma unroll
  for (int i = 0; i < 8; ++i) { r[i] = lo[i]; r[i + 8] = hi[i]; }
  return r;
}

static __device__ __forceinline__ v16h load_b16(const _Float16* W,
                                                size_t coloff, int k0, int g) {
  const _Float16* p = W + coloff + (size_t)k0 + 16 * g;
  v8h lo = *(const v8h*)(p);
  v8h hi = *(const v8h*)(p + 8);
  v16h r;
#pragma unroll
  for (int i = 0; i < 8; ++i) { r[i] = lo[i]; r[i + 8] = hi[i]; }
  return r;
}

#define WMMA_F16(a, b, c) \
  __builtin_amdgcn_wmma_f32_16x16x32_f16(false, (a), false, (b), (short)0, (c), false, false)

// ---------------------------------------------------------------------------
// CDNA5 async global->LDS copy (ASYNCcnt path); inline asm for toolchain
// portability (builtin arity differs between ROCm 7.2 and clang-23).
// VDST = LDS byte address (low 32 bits of the generic pointer = LDS offset
// per the flat->LDS aperture mapping), VADDR = 64-bit global address.
// ---------------------------------------------------------------------------
static __device__ __forceinline__ void async_copy_b128(void* lds_dst, const void* gsrc) {
  uint32_t l = (uint32_t)(uintptr_t)lds_dst;
  unsigned long long g = (unsigned long long)(uintptr_t)gsrc;
  asm volatile("global_load_async_to_lds_b128 %0, %1, off" :: "v"(l), "v"(g) : "memory");
}
static __device__ __forceinline__ void wait_async0() {
  asm volatile("s_wait_asynccnt 0x0" ::: "memory");
}

// ---------------------------------------------------------------------------
// fp32 -> f16 cast (weights)
// ---------------------------------------------------------------------------
__global__ void cast_f32_f16_kernel(const float* __restrict__ in,
                                    _Float16* __restrict__ out, int n) {
  int i = blockIdx.x * blockDim.x + threadIdx.x;
  if (i < n) out[i] = (_Float16)in[i];
}

// ---------------------------------------------------------------------------
// LayerNorm fp32 -> f16, rows padded to NPAD with zeros
// grid = B*NPAD blocks of 256
// ---------------------------------------------------------------------------
__global__ void ln_kernel(const float* __restrict__ x,
                          const float* __restrict__ gamma,
                          const float* __restrict__ beta,
                          _Float16* __restrict__ xn) {
  int row = blockIdx.x;                 // b*NPAD + n
  int n = row % NPAD;
  int b = row / NPAD;
  int tid = threadIdx.x;
  _Float16* out = xn + (size_t)row * DMODEL;
  if (n >= SEQ_N) {
    for (int d = tid; d < DMODEL; d += 256) out[d] = (_Float16)0.0f;
    return;
  }
  const float* xr = x + ((size_t)b * SEQ_N + n) * DMODEL;
  float s = 0.f, s2 = 0.f;
  for (int d = tid; d < DMODEL; d += 256) {
    float v = xr[d];
    s += v; s2 += v * v;
  }
  __shared__ float r0[256], r1[256];
  r0[tid] = s; r1[tid] = s2;
  __syncthreads();
  for (int st = 128; st > 0; st >>= 1) {
    if (tid < st) { r0[tid] += r0[tid + st]; r1[tid] += r1[tid + st]; }
    __syncthreads();
  }
  float mu  = r0[0] * (1.0f / DMODEL);
  float var = r1[0] * (1.0f / DMODEL) - mu * mu;
  float inv = rsqrtf(var + 1e-5f);
  for (int d = tid; d < DMODEL; d += 256)
    out[d] = (_Float16)((xr[d] - mu) * inv * gamma[d] + beta[d]);
}

// ---------------------------------------------------------------------------
// QKV GEMM: qkv[e, n] = sum_d xn[n,d] * Wqkv[e,d]  (f16 in, f32 acc, f16 out)
// grid = (B*M2TILES, 2304/256), block = 128 (4 waves); wave -> 32x64 tile.
// A tile (32x768, 48KB, contiguous in xn) staged once per block into LDS via
// async copy; each B fragment feeds 2 WMMAs.
// out layout: qkv[b][s][h][n][64]
// ---------------------------------------------------------------------------
__global__ void qkv_gemm_kernel(const _Float16* __restrict__ xn,
                                const _Float16* __restrict__ w,
                                _Float16* __restrict__ qkv) {
  __shared__ __align__(16) _Float16 lds_a[32 * DMODEL];

  int lane = threadIdx.x & 31;
  int wave = threadIdx.x >> 5;
  int r = lane & 15, g = lane >> 4;
  int b = blockIdx.x / M2TILES;
  int mbase = (blockIdx.x % M2TILES) * 32;
  int ebase = blockIdx.y * 256 + wave * 64;

  // stage A tile: 32 contiguous rows of xn -> LDS, VGPR-bypassing async copy
  {
    const _Float16* gbase = xn + ((size_t)b * NPAD + mbase) * DMODEL;
    for (int i = threadIdx.x; i < (32 * DMODEL) / 8; i += 128)
      async_copy_b128(lds_a + (size_t)i * 8, gbase + (size_t)i * 8);
    wait_async0();
    __syncthreads();
  }

  v8f acc[2][4] = {};
  for (int k0 = 0; k0 < DMODEL; k0 += 32) {
    v16h a0 = load_a16(lds_a, (size_t)r * DMODEL,        k0, g);
    v16h a1 = load_a16(lds_a, (size_t)(16 + r) * DMODEL, k0, g);
#pragma unroll
    for (int t = 0; t < 4; ++t) {
      size_t coloff = (size_t)(ebase + t * 16 + r) * DMODEL;
      if (k0 + 32 < DMODEL) __builtin_prefetch(w + coloff + k0 + 32, 0, 1);
      v16h bb = load_b16(w, coloff, k0, g);
      acc[0][t] = WMMA_F16(a0, bb, acc[0][t]);
      acc[1][t] = WMMA_F16(a1, bb, acc[1][t]);
    }
  }
#pragma unroll
  for (int t = 0; t < 4; ++t) {
    int e = ebase + t * 16 + r;
    int s = e / DMODEL;
    int rem = e % DMODEL;
    int h = rem >> 6;
    int dh = rem & 63;
    size_t base = ((((size_t)b * 3 + s) * NHEAD + h) * NPAD) * HD + dh;
#pragma unroll
    for (int m2 = 0; m2 < 2; ++m2) {
#pragma unroll
      for (int j = 0; j < 8; ++j) {
        int nrow = mbase + m2 * 16 + j + 8 * g;
        qkv[base + (size_t)nrow * HD] = (_Float16)acc[m2][t][j];
      }
    }
  }
}

// ---------------------------------------------------------------------------
// 2-D RoPE applied in place to q (s=0) or k (s=1); n < SEQ_N only
// ---------------------------------------------------------------------------
__global__ void rope_kernel(_Float16* __restrict__ qkv, int s) {
  int idx = blockIdx.x * blockDim.x + threadIdx.x;
  int total = NUM_B * NHEAD * SEQ_N * 32;
  if (idx >= total) return;
  int i = idx & 31;
  int n = (idx >> 5) % SEQ_N;
  int bh = idx / (32 * SEQ_N);
  int b = bh / NHEAD, h = bh % NHEAD;
  int j = i & 15;
  int half = i >> 4;
  int pos = 0;
  if (n > 0) pos = half ? ((n - 1) & 31) : ((n - 1) >> 5);
  float theta = (float)pos * __powf(10000.0f, -(float)j * (1.0f / 16.0f));
  float sn, cs;
  __sincosf(theta, &sn, &cs);
  size_t base = ((((size_t)b * 3 + s) * NHEAD + h) * NPAD + n) * HD + half * 32 + 2 * j;
  float e0 = (float)qkv[base];
  float e1 = (float)qkv[base + 1];
  qkv[base]     = (_Float16)(e0 * cs - e1 * sn);
  qkv[base + 1] = (_Float16)(e1 * cs + e0 * sn);
}

// ---------------------------------------------------------------------------
// Flash attention: one wave per (b,h, 16-query tile); key tiles of 32
// grid = (B*H, MTILES), block = 32
// ---------------------------------------------------------------------------
__global__ void flash_kernel(const _Float16* __restrict__ qkv,
                             _Float16* __restrict__ ao) {
  __shared__ __align__(16) _Float16 lds_p[16 * 32];
  __shared__ __align__(16) _Float16 lds_vt[64 * 32];

  int lane = threadIdx.x;
  int r = lane & 15, g = lane >> 4;
  int bh = blockIdx.x;
  int b = bh / NHEAD, h = bh % NHEAD;
  int qbase = blockIdx.y * 16;

  const _Float16* Q = qkv + ((((size_t)b * 3 + 0) * NHEAD + h) * NPAD) * HD;
  const _Float16* K = qkv + ((((size_t)b * 3 + 1) * NHEAD + h) * NPAD) * HD;
  const _Float16* V = qkv + ((((size_t)b * 3 + 2) * NHEAD + h) * NPAD) * HD;

  size_t qrow = (size_t)(qbase + r) * HD;
  v16h aq0 = load_a16(Q, qrow, 0, g);
  v16h aq1 = load_a16(Q, qrow, 32, g);

  v8f o[4] = {};
  float m[8], l[8];
#pragma unroll
  for (int j = 0; j < 8; ++j) { m[j] = -1e30f; l[j] = 0.0f; }
  const float scale = 0.125f;   // 1/sqrt(64)

  for (int kt = 0; kt < KTILES; ++kt) {
    int kbase = kt * 32;

    // S = Q K^T for 16x32 keys (2 C fragments, K=64 -> 2 WMMAs each)
    v8f s0 = {}, s1 = {};
    {
      size_t c0 = (size_t)(kbase + r) * HD;
      size_t c1 = (size_t)(kbase + 16 + r) * HD;
      v16h bk;
      bk = load_b16(K, c0, 0,  g); s0 = WMMA_F16(aq0, bk, s0);
      bk = load_b16(K, c0, 32, g); s0 = WMMA_F16(aq1, bk, s0);
      bk = load_b16(K, c1, 0,  g); s1 = WMMA_F16(aq0, bk, s1);
      bk = load_b16(K, c1, 32, g); s1 = WMMA_F16(aq1, bk, s1);
    }

    // stage V tile transposed into LDS: lds_vt[d][k]  (lane -> key row)
    {
      const _Float16* vrow = V + (size_t)(kbase + lane) * HD;
#pragma unroll
      for (int d = 0; d < HD; d += 8) {
        v8h vv = *(const v8h*)(vrow + d);
#pragma unroll
        for (int u = 0; u < 8; ++u) lds_vt[(d + u) * 32 + lane] = vv[u];
      }
    }

    // online softmax over the 32 columns of this key tile
    int col0 = kbase + r;
    int col1 = kbase + 16 + r;
#pragma unroll
    for (int j = 0; j < 8; ++j) {
      float x0 = (col0 < SEQ_N) ? s0[j] * scale : -1e30f;
      float x1 = (col1 < SEQ_N) ? s1[j] * scale : -1e30f;
      float mx = fmaxf(x0, x1);
      mx = fmaxf(mx, __shfl_xor(mx, 1, 32));
      mx = fmaxf(mx, __shfl_xor(mx, 2, 32));
      mx = fmaxf(mx, __shfl_xor(mx, 4, 32));
      mx = fmaxf(mx, __shfl_xor(mx, 8, 32));
      float mnew = fmaxf(m[j], mx);
      float f = __expf(m[j] - mnew);
      x0 = __expf(x0 - mnew);
      x1 = __expf(x1 - mnew);
      float rs = x0 + x1;
      rs += __shfl_xor(rs, 1, 32);
      rs += __shfl_xor(rs, 2, 32);
      rs += __shfl_xor(rs, 4, 32);
      rs += __shfl_xor(rs, 8, 32);
      l[j] = l[j] * f + rs;
      m[j] = mnew;
      int row = j + 8 * g;              // C-fragment row for this vgpr slot
      lds_p[row * 32 + r]      = (_Float16)x0;
      lds_p[row * 32 + 16 + r] = (_Float16)x1;
#pragma unroll
      for (int t = 0; t < 4; ++t) o[t][j] *= f;
    }
    __syncthreads();

    // P re-striped from LDS into A layout (row = lane%16, K = key-in-tile)
    v16h ap;
    {
      const _Float16* p = lds_p + r * 32 + 16 * g;
      v8h lo = *(const v8h*)(p);
      v8h hi = *(const v8h*)(p + 8);
#pragma unroll
      for (int i = 0; i < 8; ++i) { ap[i] = lo[i]; ap[i + 8] = hi[i]; }
    }

    // O += P * V   (B fragment from transposed LDS: contiguous over K)
#pragma unroll
    for (int t = 0; t < 4; ++t) {
      const _Float16* p = lds_vt + (size_t)(t * 16 + r) * 32 + 16 * g;
      v8h lo = *(const v8h*)(p);
      v8h hi = *(const v8h*)(p + 8);
      v16h bv;
#pragma unroll
      for (int i = 0; i < 8; ++i) { bv[i] = lo[i]; bv[i + 8] = hi[i]; }
      o[t] = WMMA_F16(ap, bv, o[t]);
    }
    __syncthreads();
  }

  // normalize and write out: ao[b][n][h*64 + d]  (f16)
#pragma unroll
  for (int j = 0; j < 8; ++j) {
    float inv = 1.0f / l[j];
    int n = qbase + j + 8 * g;
#pragma unroll
    for (int t = 0; t < 4; ++t) {
      ao[((size_t)b * NPAD + n) * DMODEL + h * HD + t * 16 + r] =
          (_Float16)(o[t][j] * inv);
    }
  }
}

// ---------------------------------------------------------------------------
// Output projection: out[b,n,e] = sum_d ao[b,n,d] * Wproj[e,d] + bias[e]
// grid = (B*M2TILES, 3), block = 128 (4 waves); wave -> 32x64 tile; A tile in
// LDS via async copy; fp32 output, padded rows skipped
// ---------------------------------------------------------------------------
__global__ void proj_gemm_kernel(const _Float16* __restrict__ ao,
                                 const _Float16* __restrict__ w,
                                 const float* __restrict__ bias,
                                 float* __restrict__ out) {
  __shared__ __align__(16) _Float16 lds_a[32 * DMODEL];

  int lane = threadIdx.x & 31;
  int wave = threadIdx.x >> 5;
  int r = lane & 15, g = lane >> 4;
  int b = blockIdx.x / M2TILES;
  int mbase = (blockIdx.x % M2TILES) * 32;
  int ebase = blockIdx.y * 256 + wave * 64;

  {
    const _Float16* gbase = ao + ((size_t)b * NPAD + mbase) * DMODEL;
    for (int i = threadIdx.x; i < (32 * DMODEL) / 8; i += 128)
      async_copy_b128(lds_a + (size_t)i * 8, gbase + (size_t)i * 8);
    wait_async0();
    __syncthreads();
  }

  v8f acc[2][4] = {};
  for (int k0 = 0; k0 < DMODEL; k0 += 32) {
    v16h a0 = load_a16(lds_a, (size_t)r * DMODEL,        k0, g);
    v16h a1 = load_a16(lds_a, (size_t)(16 + r) * DMODEL, k0, g);
#pragma unroll
    for (int t = 0; t < 4; ++t) {
      size_t coloff = (size_t)(ebase + t * 16 + r) * DMODEL;
      if (k0 + 32 < DMODEL) __builtin_prefetch(w + coloff + k0 + 32, 0, 1);
      v16h bb = load_b16(w, coloff, k0, g);
      acc[0][t] = WMMA_F16(a0, bb, acc[0][t]);
      acc[1][t] = WMMA_F16(a1, bb, acc[1][t]);
    }
  }
#pragma unroll
  for (int t = 0; t < 4; ++t) {
    int e = ebase + t * 16 + r;
    float bv = bias[e];
#pragma unroll
    for (int m2 = 0; m2 < 2; ++m2) {
#pragma unroll
      for (int j = 0; j < 8; ++j) {
        int n = mbase + m2 * 16 + j + 8 * g;
        if (n < SEQ_N)
          out[((size_t)b * SEQ_N + n) * DMODEL + e] = acc[m2][t][j] + bv;
      }
    }
  }
}

// ---------------------------------------------------------------------------
extern "C" void kernel_launch(void* const* d_in, const int* in_sizes, int n_in,
                              void* d_out, int out_size, void* d_ws, size_t ws_size,
                              hipStream_t stream) {
  (void)in_sizes; (void)n_in; (void)out_size; (void)ws_size;
  const float* x      = (const float*)d_in[0];
  const float* ln_g   = (const float*)d_in[1];
  const float* ln_b   = (const float*)d_in[2];
  const float* w_qkv  = (const float*)d_in[3];
  const float* w_proj = (const float*)d_in[4];
  const float* b_proj = (const float*)d_in[5];
  float* out = (float*)d_out;

  // workspace carve-up (256B aligned)
  char* ws = (char*)d_ws;
  size_t off = 0;
  auto carve = [&](size_t bytes) {
    char* p = ws + off;
    off += (bytes + 255) & ~(size_t)255;
    return p;
  };
  _Float16* wqkv_h  = (_Float16*)carve((size_t)3 * DMODEL * DMODEL * 2);
  _Float16* wproj_h = (_Float16*)carve((size_t)DMODEL * DMODEL * 2);
  _Float16* xn_h    = (_Float16*)carve((size_t)NUM_B * NPAD * DMODEL * 2);
  _Float16* qkv_h   = (_Float16*)carve((size_t)NUM_B * 3 * NHEAD * NPAD * HD * 2);
  _Float16* ao_h    = (_Float16*)carve((size_t)NUM_B * NPAD * DMODEL * 2);

  // 1) cast weights to f16
  {
    int n1 = 3 * DMODEL * DMODEL;
    cast_f32_f16_kernel<<<(n1 + 255) / 256, 256, 0, stream>>>(w_qkv, wqkv_h, n1);
    int n2 = DMODEL * DMODEL;
    cast_f32_f16_kernel<<<(n2 + 255) / 256, 256, 0, stream>>>(w_proj, wproj_h, n2);
  }
  // 2) LayerNorm + pad + cast
  ln_kernel<<<NUM_B * NPAD, 256, 0, stream>>>(x, ln_g, ln_b, xn_h);
  // 3) QKV GEMM
  {
    dim3 grid(NUM_B * M2TILES, (3 * DMODEL) / 256);
    qkv_gemm_kernel<<<grid, 128, 0, stream>>>(xn_h, wqkv_h, qkv_h);
  }
  // 4) RoPE on q and k
  {
    int total = NUM_B * NHEAD * SEQ_N * 32;
    int blocks = (total + 255) / 256;
    rope_kernel<<<blocks, 256, 0, stream>>>(qkv_h, 0);
    rope_kernel<<<blocks, 256, 0, stream>>>(qkv_h, 1);
  }
  // 5) flash attention
  {
    dim3 grid(NUM_B * NHEAD, MTILES);
    flash_kernel<<<grid, 32, 0, stream>>>(qkv_h, ao_h);
  }
  // 6) output projection + bias
  {
    dim3 grid(NUM_B * M2TILES, DMODEL / 256);
    proj_gemm_kernel<<<grid, 128, 0, stream>>>(ao_h, wproj_h, b_proj, out);
  }
}
